// MethodCFGEncoder_32134945308865
// MI455X (gfx1250) — matrix operand order repaired and checked by reading, not compiled, not run on hardware
//
#include <hip/hip_runtime.h>
#include <hip/hip_bf16.h>

// MethodCFGEncoder on MI455X (gfx1250, wave32).
// Gated-update GEMMs ([262144,512]x[512,256] and [8192,512]x[512,256], two
// weight matrices each) via v_wmma_f32_16x16x32_f16 with fp32 accumulation.
// M-tile per block raised to 64 rows so each packed B fragment is reused by
// 4 WMMAs (cuts L2 weight traffic 4x vs a 16-row block). Scatter duplicates
// resolved deterministically with an atomicMax tag pass.

#define N_EXPR 8192
#define MAX_TOK 32
#define DIM 256
#define KDIM 512            // 2*DIM (concat of prev|update)
#define N_APP 262144
#define N_SLOT (N_EXPR * MAX_TOK)
#define M_BLK 64            // rows per workgroup
#define CAT_PAD 8           // f16 row pad: stride 520 halves -> conflict-free b128

typedef _Float16 f16;
typedef __attribute__((ext_vector_type(16))) _Float16 v16h;
typedef __attribute__((ext_vector_type(8)))  _Float16 v8h;
typedef __attribute__((ext_vector_type(4)))  _Float16 v4h;
typedef __attribute__((ext_vector_type(8)))  float    v8f;

// ---------------------------------------------------------------------------
// Weight packing: W is [512,256] f32 row-major -> f16 WMMA B-fragment layout
// [ntile(16)][ktile(16)][lane(32)][half(16)]; each lane's fragment = 32
// contiguous bytes. B 32x16 f16, contiguous k order used here: lanes 0-15 hold K=0..15 (col N=lane),
// lanes 16-31 hold K=16..31.
// ---------------------------------------------------------------------------
__global__ void pack_weights_kernel(const float* __restrict__ Wg1,
                                    const float* __restrict__ Wc1,
                                    const float* __restrict__ Wg2,
                                    const float* __restrict__ Wc2,
                                    f16* __restrict__ dst) {
  int gid = blockIdx.x * blockDim.x + threadIdx.x;   // 0 .. 4*131072-1
  int w    = gid >> 17;
  int r    = gid & 131071;
  int h    = r & 15;
  int lane = (r >> 4) & 31;
  int kt   = (r >> 9) & 15;
  int nt   = (r >> 13) & 15;
  int n  = nt * 16 + (lane & 15);
  int kk = h + ((lane >> 4) << 4);
  int k  = kt * 32 + kk;
  const float* W = (w == 0) ? Wg1 : (w == 1) ? Wc1 : (w == 2) ? Wg2 : Wc2;
  dst[(size_t)gid] = (f16)W[k * DIM + n];
}

// ---------------------------------------------------------------------------
// Deterministic last-write arbitration for the scatter: winner = max app id.
// ---------------------------------------------------------------------------
__global__ void tag_init_kernel(int* __restrict__ tag) {
  int i = blockIdx.x * blockDim.x + threadIdx.x;
  if (i < N_SLOT) tag[i] = -1;
}

__global__ void tag_scatter_kernel(const int* __restrict__ eidx,
                                   const int* __restrict__ tidx,
                                   int* __restrict__ tag) {
  int a = blockIdx.x * blockDim.x + threadIdx.x;
  if (a < N_APP) atomicMax(&tag[eidx[a] * MAX_TOK + tidx[a]], a);
}

// ---------------------------------------------------------------------------
// WMMA inner loop for one n-tile over 4 m-tiles. B fragment loaded once per
// (kt, matrix) and reused by the 4 m-tile WMMAs. A fragment: two ds_load_b128
// per (kt, mt) from row-major padded LDS.
// A 16x32 f16 frag, contiguous k order used here (matches B packing): lanes 0-15 = rows with K in {0..15},
// lanes 16-31 = same rows with K in {16..31}.
// ---------------------------------------------------------------------------
__device__ __forceinline__ void wmma_nt(
    const f16 (*catA)[KDIM + CAT_PAD],
    const v16h* __restrict__ Bg, const v16h* __restrict__ Bc,
    int lane, int nt, v8f accF[4], v8f accC[4]) {
  int mrow = lane & 15;
  int koff = (lane & 16) ? 16 : 0;
  for (int kt = 0; kt < 16; ++kt) {
    v16h bgv = Bg[(nt * 16 + kt) * 32 + lane];
    v16h bcv = Bc[(nt * 16 + kt) * 32 + lane];
#pragma unroll
    for (int mt = 0; mt < 4; ++mt) {
      const f16* ap = &catA[mt * 16 + mrow][kt * 32 + koff];
      v8h alo = *(const v8h*)ap;
      v8h ahi = *(const v8h*)(ap + 8);
      v16h a  = __builtin_shufflevector(alo, ahi,
                  0, 1, 2, 3, 4, 5, 6, 7, 8, 9, 10, 11, 12, 13, 14, 15);
      accF[mt] = __builtin_amdgcn_wmma_f32_16x16x32_f16(
          false, a, false, bgv, (short)0, accF[mt], false, false);
      accC[mt] = __builtin_amdgcn_wmma_f32_16x16x32_f16(
          false, a, false, bcv, (short)0, accC[mt], false, false);
    }
  }
}

// ---------------------------------------------------------------------------
// Per-application gate: gather [prev_token | symbol] -> 2 GEMMs via WMMA ->
// sigmoid/relu gate -> winner-only scatter into updbuf.
// Block = 64 apps x 128 threads (4 waves); wave w owns n-tiles 4w..4w+3.
// ---------------------------------------------------------------------------
__global__ __launch_bounds__(128) void app_gate_kernel(
    const float* __restrict__ expr, const float* __restrict__ syms,
    const float* __restrict__ bg1, const float* __restrict__ bc1,
    const int* __restrict__ eidx, const int* __restrict__ tidx,
    const int* __restrict__ sidx, const int* __restrict__ tag,
    const v16h* __restrict__ Bg, const v16h* __restrict__ Bc,
    float* __restrict__ updbuf) {
  __shared__ f16 catA[M_BLK][KDIM + CAT_PAD];   // ~66.5 KB
  __shared__ int sSlot[M_BLK], sSym[M_BLK], sWin[M_BLK];

  int tid = threadIdx.x;
  int a0  = blockIdx.x * M_BLK;
  if (tid < M_BLK) {
    int app  = a0 + tid;
    int slot = eidx[app] * MAX_TOK + tidx[app];
    sSlot[tid] = slot;
    sSym[tid]  = sidx[app];
    sWin[tid]  = (tag[slot] == app) ? 1 : 0;
  }
  __syncthreads();

  // Gather: 64 rows x (64 occ quads + 64 sym quads) = 8192 float4.
  for (int j = tid; j < M_BLK * 128; j += 128) {
    int row  = j >> 7;
    int part = (j >> 6) & 1;
    int q    = j & 63;
    const float4* src = (part == 0)
        ? ((const float4*)(expr + (size_t)sSlot[row] * DIM)) + q
        : ((const float4*)(syms + (size_t)sSym[row] * DIM)) + q;
    float4 v = *src;
    v4h hv = { (f16)v.x, (f16)v.y, (f16)v.z, (f16)v.w };
    *(v4h*)&catA[row][part * DIM + q * 4] = hv;   // one ds_store_b64
  }
  __syncthreads();

  int wave = tid >> 5, lane = tid & 31;
#pragma unroll 1
  for (int i = 0; i < 4; ++i) {
    int nt = wave * 4 + i;
    v8f accF[4] = {}, accC[4] = {};
    wmma_nt(catA, Bg, Bc, lane, nt, accF, accC);
    int n = nt * 16 + (lane & 15);
    float bgv = bg1[n], bcv = bc1[n];
    int mb = (lane & 16) ? 8 : 0;          // C/D layout: hi lanes are M=r+8
#pragma unroll
    for (int mt = 0; mt < 4; ++mt) {
#pragma unroll
      for (int r = 0; r < 8; ++r) {
        int m = mt * 16 + mb + r;
        float prev = expr[(size_t)sSlot[m] * DIM + n];   // L0/L2-hot re-read
        float f = 1.0f / (1.0f + __expf(-(accF[mt][r] + bgv)));
        float c = accC[mt][r] + bcv;
        c = c > 0.0f ? c : 0.0f;
        float outv = f * prev + (1.0f - f) * c;
        if (sWin[m]) updbuf[(size_t)sSlot[m] * DIM + n] = outv;
      }
    }
  }
}

// ---------------------------------------------------------------------------
// Mean over 32 tokens, merging original rows with winner-updated rows.
// ---------------------------------------------------------------------------
__global__ void mean_kernel(const float* __restrict__ expr,
                            const float* __restrict__ updbuf,
                            const int* __restrict__ tag,
                            float* __restrict__ combined) {
  int e = blockIdx.x;
  int d = threadIdx.x;
  float s = 0.0f;
  int sb = e * MAX_TOK;
#pragma unroll 4
  for (int t = 0; t < MAX_TOK; ++t) {
    int slot = sb + t;
    const float* row = (tag[slot] >= 0) ? updbuf : expr;
    s += row[(size_t)slot * DIM + d];
  }
  combined[(size_t)e * DIM + d] = s * (1.0f / (float)MAX_TOK);
}

// ---------------------------------------------------------------------------
// CFG node gate: same WMMA GEMM over [prev_node | combined], masked store.
// Block = 64 nodes x 128 threads.
// ---------------------------------------------------------------------------
__global__ __launch_bounds__(128) void node_gate_kernel(
    const float* __restrict__ prevN, const float* __restrict__ combined,
    const float* __restrict__ bg2, const float* __restrict__ bc2,
    const unsigned char* __restrict__ mask,
    const v16h* __restrict__ Bg, const v16h* __restrict__ Bc,
    float* __restrict__ outp) {
  __shared__ f16 catA[M_BLK][KDIM + CAT_PAD];
  __shared__ int sMask[M_BLK];

  int tid = threadIdx.x;
  int e0  = blockIdx.x * M_BLK;
  if (tid < M_BLK) sMask[tid] = mask[e0 + tid] ? 1 : 0;

  for (int j = tid; j < M_BLK * 128; j += 128) {
    int row  = j >> 7;
    int part = (j >> 6) & 1;
    int q    = j & 63;
    const float4* src = (part == 0)
        ? ((const float4*)(prevN + (size_t)(e0 + row) * DIM)) + q
        : ((const float4*)(combined + (size_t)(e0 + row) * DIM)) + q;
    float4 v = *src;
    v4h hv = { (f16)v.x, (f16)v.y, (f16)v.z, (f16)v.w };
    *(v4h*)&catA[row][part * DIM + q * 4] = hv;
  }
  __syncthreads();

  int wave = tid >> 5, lane = tid & 31;
#pragma unroll 1
  for (int i = 0; i < 4; ++i) {
    int nt = wave * 4 + i;
    v8f accF[4] = {}, accC[4] = {};
    wmma_nt(catA, Bg, Bc, lane, nt, accF, accC);
    int n = nt * 16 + (lane & 15);
    float bgv = bg2[n], bcv = bc2[n];
    int mb = (lane & 16) ? 8 : 0;
#pragma unroll
    for (int mt = 0; mt < 4; ++mt) {
#pragma unroll
      for (int r = 0; r < 8; ++r) {
        int m = mt * 16 + mb + r;
        float prev = prevN[(size_t)(e0 + m) * DIM + n];
        float f = 1.0f / (1.0f + __expf(-(accF[mt][r] + bgv)));
        float c = accC[mt][r] + bcv;
        c = c > 0.0f ? c : 0.0f;
        float g = f * prev + (1.0f - f) * c;
        outp[(size_t)(e0 + m) * DIM + n] = sMask[m] ? g : prev;
      }
    }
  }
}

// ---------------------------------------------------------------------------
// Host launcher.
// Workspace layout (bytes):
//   [0, 268435456)              updbuf  f32[N_SLOT*256] (winner rows only)
//   [268435456, +1048576)       tag     i32[N_SLOT]
//   [269484032, +1048576)       packed  f16 weights: Wg1|Wc1|Wg2|Wc2
//   [270532608, +8388608)       combined f32[8192*256]
// Total ~266 MiB.
// ---------------------------------------------------------------------------
extern "C" void kernel_launch(void* const* d_in, const int* in_sizes, int n_in,
                              void* d_out, int out_size, void* d_ws, size_t ws_size,
                              hipStream_t stream) {
  const float* expr  = (const float*)d_in[0];
  const float* syms  = (const float*)d_in[1];
  const float* prevN = (const float*)d_in[2];
  const float* Wg1   = (const float*)d_in[3];
  const float* bg1   = (const float*)d_in[4];
  const float* Wc1   = (const float*)d_in[5];
  const float* bc1   = (const float*)d_in[6];
  const float* Wg2   = (const float*)d_in[7];
  const float* bg2   = (const float*)d_in[8];
  const float* Wc2   = (const float*)d_in[9];
  const float* bc2   = (const float*)d_in[10];
  const int*   eidx  = (const int*)d_in[11];
  const int*   tidx  = (const int*)d_in[12];
  const int*   sidx  = (const int*)d_in[13];
  const unsigned char* mask = (const unsigned char*)d_in[14];

  char*  ws       = (char*)d_ws;
  float* updbuf   = (float*)ws;
  int*   tag      = (int*)(ws + 268435456u);
  f16*   packed   = (f16*)(ws + 268435456u + 1048576u);
  float* combined = (float*)(ws + 268435456u + 1048576u + 1048576u);

  const v16h* pWg1 = (const v16h*)(packed);
  const v16h* pWc1 = (const v16h*)(packed + 131072);
  const v16h* pWg2 = (const v16h*)(packed + 262144);
  const v16h* pWc2 = (const v16h*)(packed + 393216);

  pack_weights_kernel<<<2048, 256, 0, stream>>>(Wg1, Wc1, Wg2, Wc2, packed);
  tag_init_kernel<<<(N_SLOT + 255) / 256, 256, 0, stream>>>(tag);
  tag_scatter_kernel<<<(N_APP + 255) / 256, 256, 0, stream>>>(eidx, tidx, tag);
  app_gate_kernel<<<N_APP / M_BLK, 128, 0, stream>>>(
      expr, syms, bg1, bc1, eidx, tidx, sidx, tag, pWg1, pWc1, updbuf);
  mean_kernel<<<N_EXPR, 256, 0, stream>>>(expr, updbuf, tag, combined);
  node_gate_kernel<<<N_EXPR / M_BLK, 128, 0, stream>>>(
      prevN, combined, bg2, bc2, mask, pWg2, pWc2, (float*)d_out);
}
